// MultiHeadDotProductAttention_57526791962887
// MI455X (gfx1250) — matrix operand order repaired and checked by reading, not compiled, not run on hardware
//
#include <hip/hip_runtime.h>

#define B_BATCH 2
#define N_SEQ   2048
#define D_MODEL 1024
#define H_HEADS 16
#define HD      64
#define NEG_BIG (-1.0e10f)

typedef __attribute__((ext_vector_type(16))) __bf16 v16bf;
typedef __attribute__((ext_vector_type(8)))  __bf16 v8bf;
typedef __attribute__((ext_vector_type(8)))  float  v8f;

// D = A(16x32 bf16) * B(32x16 bf16) + C(16x16 f32)
__device__ __forceinline__ v8f wmma_bf16(v16bf a, v16bf b, v8f c) {
  return __builtin_amdgcn_wmma_f32_16x16x32_bf16(false, a, false, b, (short)0, c,
                                                 false, false);
}

// A fragment 16x32 (MxK), row-major source, row = lane%16.
// ISA layout: lanes 0-15 hold K = k0+0..7 and k0+16..23; lanes 16-31 hold +8.
// -> two contiguous 16B loads per lane.
__device__ __forceinline__ v16bf load_frag_a(const __bf16* base, int stride,
                                             int k0, int lane) {
  const int m = lane & 15, half = lane >> 4;
  const __bf16* p = base + (size_t)m * stride + k0 + half * 8;
  v8bf lo = *(const v8bf*)(p);
  v8bf hi = *(const v8bf*)(p + 16);
  v16bf r;
#pragma unroll
  for (int i = 0; i < 8; ++i) { r[i] = lo[i]; r[i + 8] = hi[i]; }
  return r;
}

// B fragment 32x16 (KxN) taken from a [n][k] (transposed) source.
// ISA layout: lane holds col n=lane%16, K = k0+8*(lane/16)+0..7 and +16..23
// -> two contiguous 16B loads per lane.
__device__ __forceinline__ v16bf load_frag_b(const __bf16* baseT, int stride,
                                             int n0, int k0, int lane) {
  const int n = lane & 15, half = lane >> 4;
  const __bf16* p = baseT + (size_t)(n0 + n) * stride + k0 + half * 8;
  v8bf lo = *(const v8bf*)(p);
  v8bf hi = *(const v8bf*)(p + 16);
  v16bf r;
#pragma unroll
  for (int i = 0; i < 8; ++i) { r[i] = lo[i]; r[i + 8] = hi[i]; }
  return r;
}

// ---------------- conversion kernels ----------------
__global__ void k_cvt(const float* __restrict__ in, __bf16* __restrict__ out, int n) {
  int i = blockIdx.x * blockDim.x + threadIdx.x;
  if (i < n) out[i] = (__bf16)in[i];
}

// in[R][C] f32 -> out[C][R] bf16
__global__ void k_tr(const float* __restrict__ in, __bf16* __restrict__ out,
                     int R, int C) {
  int i = blockIdx.x * blockDim.x + threadIdx.x;
  if (i < R * C) {
    int c = i / R, k = i - c * R;
    out[i] = (__bf16)in[(size_t)k * C + c];
  }
}

// ---------------- QKV projection GEMM ----------------
// X[4096][1024] bf16 @ Wqkv (via WqkvT[3072][1024] bf16) -> scatter Q,K,V
// One wave computes a 32x64 tile: 2 A-frags reuse each B-frag twice.
__global__ __launch_bounds__(128) void k_gemm_qkv(const __bf16* __restrict__ X,
                                                  const __bf16* __restrict__ Wt,
                                                  __bf16* __restrict__ Qb,
                                                  __bf16* __restrict__ Kb,
                                                  __bf16* __restrict__ Vt) {
  const int lane = threadIdx.x & 31;
  const int wid  = blockIdx.x * 4 + (threadIdx.x >> 5);
  const int mt = wid / 48, ct = wid % 48;     // 128 x 48 tiles of 32x64
  const int row0 = mt * 32, col0 = ct * 64;

  v8f acc[2][4];
#pragma unroll
  for (int r = 0; r < 2; ++r)
#pragma unroll
    for (int c = 0; c < 4; ++c) acc[r][c] = (v8f){};

  for (int k0 = 0; k0 < D_MODEL; k0 += 32) {
    const v16bf a0 = load_frag_a(X + (size_t)row0 * D_MODEL,        D_MODEL, k0, lane);
    const v16bf a1 = load_frag_a(X + (size_t)(row0 + 16) * D_MODEL, D_MODEL, k0, lane);
#pragma unroll
    for (int c = 0; c < 4; ++c) {
      const v16bf b = load_frag_b(Wt, D_MODEL, col0 + 16 * c, k0, lane);
      acc[0][c] = wmma_bf16(a0, b, acc[0][c]);
      acc[1][c] = wmma_bf16(a1, b, acc[1][c]);
    }
  }

  const int s = col0 >> 10;               // 0=q 1=k 2=v
  const int h = (col0 & 1023) >> 6;
  const int half = lane >> 4, nl = lane & 15;
#pragma unroll
  for (int r = 0; r < 2; ++r)
#pragma unroll
    for (int j = 0; j < 8; ++j) {
      const int grow = row0 + 16 * r + j + 8 * half;
      const int bb = grow >> 11, n = grow & (N_SEQ - 1);
      const size_t bh = (size_t)(bb * H_HEADS + h);
#pragma unroll
      for (int c = 0; c < 4; ++c) {
        const int d = 16 * c + nl;
        const __bf16 val = (__bf16)acc[r][c][j];
        if (s == 0)      Qb[(bh * N_SEQ + n) * HD + d] = val;
        else if (s == 1) Kb[(bh * N_SEQ + n) * HD + d] = val;
        else             Vt[(bh * HD + d) * N_SEQ + n] = val;   // V transposed
      }
    }
}

// ---------------- flash attention ----------------
// One wave per (b, h, 32-row q tile). Online softmax over 64-col key chunks.
// K/V B-fragments are each reused by both 16-row q fragments.
__global__ __launch_bounds__(128) void k_attn(const __bf16* __restrict__ Qb,
                                              const __bf16* __restrict__ Kb,
                                              const __bf16* __restrict__ Vt,
                                              const int*    __restrict__ mask,
                                              __bf16*       __restrict__ Xo) {
  __shared__ __align__(16) __bf16 Pbuf[4][32 * 72];   // per-wave 32x64 probs, pad 72
  const int lane = threadIdx.x & 31;
  const int wave = threadIdx.x >> 5;
  const int tile = blockIdx.x * 4 + wave;             // 2048 tiles total
  const int b  = tile >> 10;
  const int h  = (tile >> 6) & 15;
  const int qt = tile & 63;
  const int qbase = qt * 32;
  const int half = lane >> 4, nl = lane & 15;

  const __bf16* Qh = Qb + (size_t)(b * H_HEADS + h) * N_SEQ * HD;
  const __bf16* Kh = Kb + (size_t)(b * H_HEADS + h) * N_SEQ * HD;
  const __bf16* Vh = Vt + (size_t)(b * H_HEADS + h) * HD * N_SEQ; // [hd][seq]
  __bf16* P = &Pbuf[wave][0];

  v16bf qa[2][2];
#pragma unroll
  for (int r = 0; r < 2; ++r) {
    qa[r][0] = load_frag_a(Qh + (size_t)(qbase + 16 * r) * HD, HD, 0,  lane);
    qa[r][1] = load_frag_a(Qh + (size_t)(qbase + 16 * r) * HD, HD, 32, lane);
  }

  v8f o[2][4];
  float mrow[2][8], lrow[2][8];
#pragma unroll
  for (int r = 0; r < 2; ++r) {
#pragma unroll
    for (int c = 0; c < 4; ++c) o[r][c] = (v8f){};
#pragma unroll
    for (int j = 0; j < 8; ++j) { mrow[r][j] = -3.0e38f; lrow[r][j] = 0.f; }
  }

  for (int kb = 0; kb < N_SEQ; kb += 64) {
    if (kb + 64 < N_SEQ) {                       // global_prefetch_b8 next chunk
      __builtin_prefetch(Kh + (size_t)(kb + 64) * HD, 0, 1);
      __builtin_prefetch(Vh + (kb + 64), 0, 1);
    }
    // ----- scores: 2x4 tiles of 16x16, K-dim = hd = 64 (two K=32 steps) -----
    v8f s[2][4];
#pragma unroll
    for (int t = 0; t < 4; ++t) {
      const v16bf bk0 = load_frag_b(Kh, HD, kb + 16 * t, 0,  lane);
      const v16bf bk1 = load_frag_b(Kh, HD, kb + 16 * t, 32, lane);
#pragma unroll
      for (int r = 0; r < 2; ++r) {
        v8f c = {};
        c = wmma_bf16(qa[r][0], bk0, c);
        c = wmma_bf16(qa[r][1], bk1, c);
        s[r][t] = c;
      }
    }
#pragma unroll
    for (int r = 0; r < 2; ++r) {
      // ----- mask bias + per-lane partial row max -----
      float tmax[8];
#pragma unroll
      for (int j = 0; j < 8; ++j) {
        const int qr = qbase + 16 * r + j + 8 * half;
        const int* mp = mask + (size_t)qr * N_SEQ + kb + nl;
        float mx = -3.0e38f;
#pragma unroll
        for (int t = 0; t < 4; ++t) {
          const float sv = s[r][t][j] + ((mp[16 * t] <= 0) ? NEG_BIG : 0.f);
          s[r][t][j] = sv;
          mx = fmaxf(mx, sv);
        }
        tmax[j] = mx;
      }
      // cross-lane row max (rows live in 16-lane halves; xor<16 stays in half)
#pragma unroll
      for (int j = 0; j < 8; ++j) {
#pragma unroll
        for (int d = 1; d < 16; d <<= 1)
          tmax[j] = fmaxf(tmax[j], __shfl_xor(tmax[j], d, 32));
      }
      float scal[8];
#pragma unroll
      for (int j = 0; j < 8; ++j) {
        const float nm = fmaxf(mrow[r][j], tmax[j]);
        scal[j] = __expf(mrow[r][j] - nm);
        mrow[r][j] = nm;
      }
      // ----- exponentiate, stage probs to LDS (C-layout -> A-layout) -----
      float rsum[8];
#pragma unroll
      for (int j = 0; j < 8; ++j) {
        float rs = 0.f;
#pragma unroll
        for (int t = 0; t < 4; ++t) {
          const float p = __expf(s[r][t][j] - mrow[r][j]);
          rs += p;
          P[(16 * r + j + 8 * half) * 72 + 16 * t + nl] = (__bf16)p;
        }
        rsum[j] = rs;
      }
#pragma unroll
      for (int j = 0; j < 8; ++j) {
#pragma unroll
        for (int d = 1; d < 16; d <<= 1)
          rsum[j] += __shfl_xor(rsum[j], d, 32);
        lrow[r][j] = lrow[r][j] * scal[j] + rsum[j];
      }
      // rescale running output
#pragma unroll
      for (int c = 0; c < 4; ++c)
#pragma unroll
        for (int j = 0; j < 8; ++j) o[r][c][j] *= scal[j];
    }
    // ----- P @ V : A = probs 32x64 (2x2 frags), B-frags reused by both rows -----
    v16bf pa[2][2];
#pragma unroll
    for (int r = 0; r < 2; ++r) {
      pa[r][0] = load_frag_a(P + 16 * r * 72, 72, 0,  lane);
      pa[r][1] = load_frag_a(P + 16 * r * 72, 72, 32, lane);
    }
#pragma unroll
    for (int kc = 0; kc < 2; ++kc)
#pragma unroll
      for (int nt = 0; nt < 4; ++nt) {
        const v16bf bv = load_frag_b(Vh, N_SEQ, 16 * nt, kb + 32 * kc, lane);
        o[0][nt] = wmma_bf16(pa[0][kc], bv, o[0][nt]);
        o[1][nt] = wmma_bf16(pa[1][kc], bv, o[1][nt]);
      }
  }

  // ----- normalize + store bf16 into [b][n][h*64+d] -----
#pragma unroll
  for (int r = 0; r < 2; ++r)
#pragma unroll
    for (int j = 0; j < 8; ++j) {
      const float inv = 1.f / lrow[r][j];
      const int qr = qbase + 16 * r + j + 8 * half;
      __bf16* op = Xo + (size_t)(b * N_SEQ + qr) * (H_HEADS * HD) + h * HD + nl;
#pragma unroll
      for (int nt = 0; nt < 4; ++nt)
        op[16 * nt] = (__bf16)(o[r][nt][j] * inv);
    }
}

// ---------------- output projection GEMM ----------------
// Xo[4096][1024] bf16 @ W_out (via WoutT[1024][1024]) -> f32 out[4096][1024]
__global__ __launch_bounds__(128) void k_gemm_out(const __bf16* __restrict__ Xo,
                                                  const __bf16* __restrict__ Wt,
                                                  float* __restrict__ out) {
  const int lane = threadIdx.x & 31;
  const int wid  = blockIdx.x * 4 + (threadIdx.x >> 5);
  const int mt = wid / 16, ct = wid % 16;     // 128 x 16 tiles of 32x64
  const int row0 = mt * 32, col0 = ct * 64;

  v8f acc[2][4];
#pragma unroll
  for (int r = 0; r < 2; ++r)
#pragma unroll
    for (int c = 0; c < 4; ++c) acc[r][c] = (v8f){};

  for (int k0 = 0; k0 < D_MODEL; k0 += 32) {
    const v16bf a0 = load_frag_a(Xo + (size_t)row0 * D_MODEL,        D_MODEL, k0, lane);
    const v16bf a1 = load_frag_a(Xo + (size_t)(row0 + 16) * D_MODEL, D_MODEL, k0, lane);
#pragma unroll
    for (int c = 0; c < 4; ++c) {
      const v16bf b = load_frag_b(Wt, D_MODEL, col0 + 16 * c, k0, lane);
      acc[0][c] = wmma_bf16(a0, b, acc[0][c]);
      acc[1][c] = wmma_bf16(a1, b, acc[1][c]);
    }
  }

  const int half = lane >> 4, nl = lane & 15;
#pragma unroll
  for (int r = 0; r < 2; ++r)
#pragma unroll
    for (int j = 0; j < 8; ++j) {
      const int grow = row0 + 16 * r + j + 8 * half;
      float* op = out + (size_t)grow * D_MODEL + col0 + nl;
#pragma unroll
      for (int c = 0; c < 4; ++c) op[16 * c] = acc[r][c][j];
    }
}

// ---------------- launcher ----------------
extern "C" void kernel_launch(void* const* d_in, const int* in_sizes, int n_in,
                              void* d_out, int out_size, void* d_ws, size_t ws_size,
                              hipStream_t stream) {
  const float* inputs_q = (const float*)d_in[0];
  const int*   mask     = (const int*)d_in[1];
  const float* W_qkv    = (const float*)d_in[2];
  const float* W_out    = (const float*)d_in[3];
  float* out = (float*)d_out;

  char* ws = (char*)d_ws;
  size_t off = 0;
  const size_t nX  = (size_t)B_BATCH * N_SEQ * D_MODEL;        // 4M elems
  const size_t nHd = (size_t)B_BATCH * H_HEADS * N_SEQ * HD;   // 4M elems

  __bf16* Xb    = (__bf16*)(ws + off); off += nX * 2;                            // 8 MB
  __bf16* WqkvT = (__bf16*)(ws + off); off += (size_t)3 * D_MODEL * D_MODEL * 2; // 6 MB
  __bf16* WoutT = (__bf16*)(ws + off); off += (size_t)D_MODEL * D_MODEL * 2;     // 2 MB
  __bf16* Qb    = (__bf16*)(ws + off); off += nHd * 2;                           // 8 MB
  __bf16* Kb    = (__bf16*)(ws + off); off += nHd * 2;                           // 8 MB
  __bf16* Vt    = (__bf16*)(ws + off); off += nHd * 2;                           // 8 MB
  __bf16* Xo    = (__bf16*)(ws + off); off += nX * 2;                            // 8 MB

  // 1) convert activations + transpose/convert weights to bf16
  k_cvt<<<(int)(nX / 256), 256, 0, stream>>>(inputs_q, Xb, (int)nX);
  k_tr <<<(3 * D_MODEL * D_MODEL) / 256, 256, 0, stream>>>(W_qkv, WqkvT,
                                                           D_MODEL, 3 * D_MODEL);
  k_tr <<<(D_MODEL * D_MODEL) / 256, 256, 0, stream>>>(W_out, WoutT,
                                                       D_MODEL, D_MODEL);
  // 2) QKV projection (128*48 wave-tiles of 32x64, 4 waves/block)
  k_gemm_qkv<<<1536, 128, 0, stream>>>(Xb, WqkvT, Qb, Kb, Vt);
  // 3) flash attention (2048 wave-tiles of 32 q-rows)
  k_attn<<<512, 128, 0, stream>>>(Qb, Kb, Vt, mask, Xo);
  // 4) output projection (128*16 wave-tiles of 32x64)
  k_gemm_out<<<512, 128, 0, stream>>>(Xo, WoutT, out);
}
